// SphereTracingRenderer_46282567581995
// MI455X (gfx1250) — matrix operand order repaired and checked by reading, not compiled, not run on hardware
//
#include <hip/hip_runtime.h>
#include <hip/hip_bf16.h>
#include <math.h>

#define NEAR_T   0.8f
#define FAR_T    6.0f
#define MAX_IT   32
#define EPS_T    1e-5f
#define NRAYS    16384
#define HID      256

typedef __attribute__((ext_vector_type(16))) __bf16 v16bf;
typedef __attribute__((ext_vector_type(8)))  float  v8f;

union BFrag {
    float4 f[2];
    v16bf  v;
};

__device__ __forceinline__ void wait_ds0() {
#if __has_builtin(__builtin_amdgcn_s_wait_dscnt)
    __builtin_amdgcn_s_wait_dscnt(0);
#else
    asm volatile("s_wait_dscnt 0" ::: "memory");
#endif
}

struct Smem {
    __bf16 W2T[HID * HID];   // W2 transposed: [n][k], bf16, 128 KB
    float  W1[3 * HID];      // [i][k] row-major (3 rows of 256)
    float  b1[HID];
    float  b2[HID];
    float  W3[HID];
    float  Wc1[3 * HID];
    float  bc1[HID];
    float  Wc2[HID * 3];     // [k][c] row-major
    float  sbuf[8 * 32];     // per-wave SDF exchange
};

__global__ __launch_bounds__(256) void sphere_trace_wmma(
    const float* __restrict__ origins, const float* __restrict__ dirs,
    const float* __restrict__ gW1, const float* __restrict__ gb1,
    const float* __restrict__ gW2, const float* __restrict__ gb2,
    const float* __restrict__ gW3, const float* __restrict__ gb3,
    const float* __restrict__ gWc1, const float* __restrict__ gbc1,
    const float* __restrict__ gWc2, const float* __restrict__ gbc2,
    float* __restrict__ out)
{
    __shared__ Smem sm;
    const int tid = threadIdx.x;

    // ---- Stage W2 -> LDS as bf16 transposed [n][k] (coalesced f32x4 reads) ----
    for (int idx = tid * 4; idx < HID * HID; idx += 256 * 4) {
        float4 v = *(const float4*)(gW2 + idx);       // row-major [k][n]
        int k = idx >> 8;
        int n = idx & 255;
        sm.W2T[(n + 0) * HID + k] = (__bf16)v.x;
        sm.W2T[(n + 1) * HID + k] = (__bf16)v.y;
        sm.W2T[(n + 2) * HID + k] = (__bf16)v.z;
        sm.W2T[(n + 3) * HID + k] = (__bf16)v.w;
    }
    for (int i = tid; i < 3 * HID; i += 256) {
        sm.W1[i]  = gW1[i];
        sm.Wc1[i] = gWc1[i];
        sm.Wc2[i] = gWc2[i];
    }
    for (int i = tid; i < HID; i += 256) {
        sm.b1[i]  = gb1[i];
        sm.b2[i]  = gb2[i];
        sm.W3[i]  = gW3[i];
        sm.bc1[i] = gbc1[i];
    }
    __syncthreads();

    // ---- Per-lane ray state: lane L owns ray rayBase + L ----
    const int lane = tid & 31;
    const int wave = tid >> 5;
    const int ray  = (blockIdx.x * 8 + wave) * 32 + lane;
    const int hi   = lane >> 4;      // which 16-lane half
    const int col  = lane & 15;      // row/col index within tile
    float* sbuf = sm.sbuf + wave * 32;

    const float ox = origins[ray * 3 + 0], oy = origins[ray * 3 + 1], oz = origins[ray * 3 + 2];
    const float dx = dirs[ray * 3 + 0],    dy = dirs[ray * 3 + 1],    dz = dirs[ray * 3 + 2];
    const float sx = fmaf(NEAR_T, dx, ox), sy = fmaf(NEAR_T, dy, oy), sz = fmaf(NEAR_T, dz, oz);
    const float b3v = gb3[0];

    float t = 0.0f;
    int active = 1;

    for (int it = 0; it < MAX_IT; ++it) {
        const float px = fmaf(t, dx, sx);
        const float py = fmaf(t, dy, sy);
        const float pz = fmaf(t, dz, sz);

        // ---- Layer 1 (K=3) computed straight into A-fragment layout ----
        // A tile tt covers rays base+16*tt .. +15; lane holds row m = col.
        v16bf afrag[2][8];
#pragma unroll
        for (int tt = 0; tt < 2; ++tt) {
            const int src = tt * 16 + col;
            const float qx = __shfl(px, src, 32);
            const float qy = __shfl(py, src, 32);
            const float qz = __shfl(pz, src, 32);
#pragma unroll
            for (int kf = 0; kf < 8; ++kf) {
#pragma unroll
                for (int e = 0; e < 16; ++e) {
                    // contiguous A layout: lane half selects +16; elems contiguous in K
                    const int K = 32 * kf + 16 * hi + e;
                    float v = fmaf(qx, sm.W1[K],
                              fmaf(qy, sm.W1[256 + K],
                              fmaf(qz, sm.W1[512 + K], sm.b1[K])));
                    afrag[tt][kf][e] = (__bf16)fmaxf(v, 0.0f);
                }
            }
        }

        // ---- Layer 2 (256x256) on WMMA (C starts at inline 0; b2 added in
        //      the epilogue), layer 3 (N=1) fused via W3. N-tile loop NOT
        //      unrolled to keep VGPR count under 256. ----
        float sp0[8], sp1[8];
#pragma unroll
        for (int r = 0; r < 8; ++r) { sp0[r] = 0.0f; sp1[r] = 0.0f; }

#pragma unroll 1
        for (int nt = 0; nt < 16; ++nt) {
            const int n = nt * 16 + col;            // this lane's output column
            const float b2v = sm.b2[n];
            const float w3v = sm.W3[n];
            const __bf16* bbase = sm.W2T + n * HID + 16 * hi;  // 32B aligned

            v8f acc0 = {}, acc1 = {};               // inline-0 C operand
            BFrag cur, nxt;
            cur.f[0] = ((const float4*)(bbase))[0];
            cur.f[1] = ((const float4*)(bbase))[1];
#pragma unroll
            for (int kf = 0; kf < 8; ++kf) {
                if (kf < 7) {                       // prefetch next K-fragment
                    const float4* np = (const float4*)(bbase + 32 * (kf + 1));
                    nxt.f[0] = np[0];
                    nxt.f[1] = np[1];
                }
                acc0 = __builtin_amdgcn_wmma_f32_16x16x32_bf16(
                           false, afrag[0][kf], false, cur.v, (short)0, acc0, false, false);
                acc1 = __builtin_amdgcn_wmma_f32_16x16x32_bf16(
                           false, afrag[1][kf], false, cur.v, (short)0, acc1, false, false);
                cur = nxt;
            }
#pragma unroll
            for (int r = 0; r < 8; ++r) {
                sp0[r] = fmaf(fmaxf(acc0[r] + b2v, 0.0f), w3v, sp0[r]);
                sp1[r] = fmaf(fmaxf(acc1[r] + b2v, 0.0f), w3v, sp1[r]);
            }
        }

        // ---- Reduce over the 16 lanes of each half (columns of the D tile) ----
#pragma unroll
        for (int r = 0; r < 8; ++r) {
#pragma unroll
            for (int m = 1; m < 16; m <<= 1) {
                sp0[r] += __shfl_xor(sp0[r], m, 32);
                sp1[r] += __shfl_xor(sp1[r], m, 32);
            }
        }
        // Lane with col==0 in each half publishes rows m = 8*hi + r
        if (col == 0) {
#pragma unroll
            for (int r = 0; r < 8; ++r) {
                sbuf[0 * 16 + 8 * hi + r] = sp0[r];
                sbuf[1 * 16 + 8 * hi + r] = sp1[r];
            }
        }
        wait_ds0();
        const float s = sbuf[lane] + b3v;     // sbuf[tile*16+m] == sbuf[lane]

        // ---- Masked sphere-tracing step (uniform control flow) ----
        const float pn = sqrtf(fmaf(px, px, fmaf(py, py, pz * pz)));
        const int cond = active && (s > EPS_T) && (pn < FAR_T);
        t = cond ? (t + s) : t;
        active = cond;
    }

    // ---- Color head: 3 -> 256 -> 3 + sigmoid, fused in VALU ----
    const float px = fmaf(t, dx, sx);
    const float py = fmaf(t, dy, sy);
    const float pz = fmaf(t, dz, sz);
    const float ex = fmaf(FAR_T, dx, ox);
    const float ey = fmaf(FAR_T, dy, oy);
    const float ez = fmaf(FAR_T, dz, oz);
    const bool hit = (px != ex) || (py != ey) || (pz != ez);

    float cr = 0.0f, cg = 0.0f, cb = 0.0f;
#pragma unroll 4
    for (int k = 0; k < HID; ++k) {
        float h = fmaf(px, sm.Wc1[k],
                  fmaf(py, sm.Wc1[256 + k],
                  fmaf(pz, sm.Wc1[512 + k], sm.bc1[k])));
        h = fmaxf(h, 0.0f);
        cr = fmaf(h, sm.Wc2[k * 3 + 0], cr);
        cg = fmaf(h, sm.Wc2[k * 3 + 1], cg);
        cb = fmaf(h, sm.Wc2[k * 3 + 2], cb);
    }
    cr = 1.0f / (1.0f + __expf(-(cr + gbc2[0])));
    cg = 1.0f / (1.0f + __expf(-(cg + gbc2[1])));
    cb = 1.0f / (1.0f + __expf(-(cb + gbc2[2])));

    out[ray * 3 + 0] = hit ? cr : 0.0f;
    out[ray * 3 + 1] = hit ? cg : 0.0f;
    out[ray * 3 + 2] = hit ? cb : 0.0f;
}

extern "C" void kernel_launch(void* const* d_in, const int* in_sizes, int n_in,
                              void* d_out, int out_size, void* d_ws, size_t ws_size,
                              hipStream_t stream) {
    (void)in_sizes; (void)n_in; (void)d_ws; (void)ws_size; (void)out_size;
    const float* origins = (const float*)d_in[0];
    const float* dirs    = (const float*)d_in[1];
    const float* W1      = (const float*)d_in[2];
    const float* b1      = (const float*)d_in[3];
    const float* W2      = (const float*)d_in[4];
    const float* b2      = (const float*)d_in[5];
    const float* W3      = (const float*)d_in[6];
    const float* b3      = (const float*)d_in[7];
    const float* Wc1     = (const float*)d_in[8];
    const float* bc1     = (const float*)d_in[9];
    const float* Wc2     = (const float*)d_in[10];
    const float* bc2     = (const float*)d_in[11];
    float* out = (float*)d_out;

    // 32 rays per wave, 8 waves per block -> 256 rays/block, 64 blocks
    dim3 grid(NRAYS / 256), block(256);
    sphere_trace_wmma<<<grid, block, 0, stream>>>(
        origins, dirs, W1, b1, W2, b2, W3, b3, Wc1, bc1, Wc2, bc2, out);
}
